// _ECE_criterion_54494545052055
// MI455X (gfx1250) — compile-verified, matrix operand order checked by reading, not run on hardware
//
#include <hip/hip_runtime.h>
#include <hip/hip_bf16.h>

typedef __attribute__((ext_vector_type(16))) _Float16 v16h;
typedef __attribute__((ext_vector_type(8)))  float    v8f;

#define NBINS 10

// ---------------------------------------------------------------------------
// Main kernel: one wave processes 32 rows per iteration (lane = row).
// Per-chunk bin accumulation is done with 3 WMMA ops:
//   A (16x32 f16, documented layout)  x  B (32x16 f16, all ones)
//   -> D[m][n] = sum_k A[m][k]  for every column n  (B-layout agnostic)
// A_cnt[m][k] = (bin_k == m)                     (from 10 ballots)
// A_acc[m][k] = (bin_k == m && correct_k)        (ballots + AND)
// A_del[m][k] = (bin_k == m) ? delta_k : 0       (ds_bpermute gather)
// ---------------------------------------------------------------------------
__global__ __launch_bounds__(256) void ece_main(const float* __restrict__ logits,
                                                const int*   __restrict__ labels,
                                                float*       __restrict__ partials,
                                                int N)
{
    __shared__ float sbins[3 * NBINS];
    const int tid = threadIdx.x;
    if (tid < 3 * NBINS) sbins[tid] = 0.0f;
    __syncthreads();

    const int lane         = tid & 31;
    const int wavesPerBlk  = blockDim.x >> 5;
    const int waveId       = blockIdx.x * wavesPerBlk + (tid >> 5);
    const int nWaves       = gridDim.x * wavesPerBlk;
    const int m            = lane & 15;              // one-hot row this lane holds
    const int g8           = (lane >= 16) ? 8 : 0;   // D rows 8..15 live in lanes 16..31
    const int kHiOff       = (lane >= 16) ? 8 : 0;   // A-layout K offset per lane group

    v16h ones;
    #pragma unroll
    for (int s = 0; s < 16; ++s) ones[s] = (_Float16)1.0f;

    v8f accCnt = {};
    v8f accAcc = {};
    v8f accDel = {};

    const int step = nWaves * 32;
    for (int base = waveId * 32; base < N; base += step) {
        const int r = base + lane;

        int   bin     = 100;     // sentinel: matches no ballot bucket
        int   correct = 0;
        float delta   = 0.0f;

        // Prefetch next streaming chunk (global_prefetch_b8).
        {
            const int nr = r + step;
            if (nr < N) __builtin_prefetch(logits + (size_t)nr * 32, 0, 0);
        }

        if (r < N) {
            const float4* row = (const float4*)(logits + (size_t)r * 32);
            float best = -3.4e38f;
            int   bidx = 0;
            #pragma unroll
            for (int j = 0; j < 8; ++j) {       // 8 x global_load_b128, 128B/row
                float4 v = row[j];
                if (v.x > best) { best = v.x; bidx = j * 4 + 0; }
                if (v.y > best) { best = v.y; bidx = j * 4 + 1; }
                if (v.z > best) { best = v.z; bidx = j * 4 + 2; }
                if (v.w > best) { best = v.w; bidx = j * 4 + 3; }
            }
            // sigmoid monotonic: conf = sigmoid(max logit), pred = argmax(logits)
            const float conf = 1.0f / (1.0f + __expf(-best));
            int b = (int)ceilf(conf * 10.0f) - 1;        // half-open bins (l, u]
            bin   = b < 0 ? 0 : (b > NBINS - 1 ? NBINS - 1 : b);
            delta = conf - 0.1f * (float)bin;            // in (0, ~0.1]; f16-friendly
            correct = (bidx == labels[r]) ? 1 : 0;
        }

        // --- ballots: full-wave, uniform control flow (EXEC all ones) -------
        unsigned myMask = 0u;
        const unsigned accBallot = __builtin_amdgcn_ballot_w32(correct != 0);
        #pragma unroll
        for (int q = 0; q < NBINS; ++q) {
            const unsigned bal = __builtin_amdgcn_ballot_w32(bin == q);
            if (m == q) myMask = bal;
        }
        const unsigned myAccMask = myMask & accBallot;

        // pack this lane's delta as f16 for cross-lane gather
        const _Float16 dh = (_Float16)delta;
        const int dpack = (int)__builtin_bit_cast(unsigned short, dh);

        // --- build A matrices per documented 16-bit A layout (ISA 7.12.2) ---
        // lane<16 : V0..3 = K 0..7,  V4..7 = K 16..23
        // lane>=16: V0..3 = K 8..15, V4..7 = K 24..31
        v16h aCnt, aAcc, aDel;
        #pragma unroll
        for (int s = 0; s < 16; ++s) {
            const int k = (s & 7) + kHiOff + ((s >= 8) ? 16 : 0);
            const unsigned hit = (myMask >> k) & 1u;
            aCnt[s] = hit ? (_Float16)1.0f : (_Float16)0.0f;
            aAcc[s] = ((myAccMask >> k) & 1u) ? (_Float16)1.0f : (_Float16)0.0f;
            const int got = __builtin_amdgcn_ds_bpermute(k << 2, dpack);
            const _Float16 dv =
                __builtin_bit_cast(_Float16, (unsigned short)(got & 0xFFFF));
            aDel[s] = hit ? dv : (_Float16)0.0f;
        }

        // --- 3x v_wmma_f32_16x16x32_f16, f32 accumulate ---------------------
        accCnt = __builtin_amdgcn_wmma_f32_16x16x32_f16(
            false, aCnt, false, ones, (short)0, accCnt, false, false);
        accAcc = __builtin_amdgcn_wmma_f32_16x16x32_f16(
            false, aAcc, false, ones, (short)0, accAcc, false, false);
        accDel = __builtin_amdgcn_wmma_f32_16x16x32_f16(
            false, aDel, false, ones, (short)0, accDel, false, false);
    }

    // Extract column N=0 of each D: VGPR g, lane 0 -> M=g ; lane 16 -> M=g+8.
    if ((lane & 15) == 0) {
        #pragma unroll
        for (int g = 0; g < 8; ++g) {
            const int b = g8 + g;
            if (b < NBINS) {
                atomicAdd(&sbins[b],             accCnt[g]);
                atomicAdd(&sbins[NBINS + b],     accAcc[g]);
                atomicAdd(&sbins[2 * NBINS + b], accDel[g]);
            }
        }
    }
    __syncthreads();

    if (tid < 3 * NBINS) partials[(size_t)blockIdx.x * 32 + tid] = sbins[tid];
}

// ---------------------------------------------------------------------------
// Deterministic finalize: fixed-order reduction over block partials, then ECE.
// ---------------------------------------------------------------------------
__global__ __launch_bounds__(32) void ece_finalize(const float* __restrict__ partials,
                                                   int nblocks, float inv_n,
                                                   float* __restrict__ out)
{
    __shared__ float tot[3 * NBINS];
    const int j = threadIdx.x;
    if (j < 3 * NBINS) {
        float s = 0.0f;
        for (int b = 0; b < nblocks; ++b) s += partials[(size_t)b * 32 + j];
        tot[j] = s;
    }
    __syncthreads();
    if (j == 0) {
        float ece = 0.0f;
        for (int b = 0; b < NBINS; ++b) {
            const float cnt  = tot[b];
            const float sacc = tot[NBINS + b];
            const float sdel = tot[2 * NBINS + b];
            if (cnt > 0.0f) {
                const float sconf = cnt * (0.1f * (float)b) + sdel; // undo delta split
                const float gap   = fabsf(sconf / cnt - sacc / cnt);
                ece += gap * (cnt * inv_n);
            }
        }
        out[0] = ece;
    }
}

extern "C" void kernel_launch(void* const* d_in, const int* in_sizes, int n_in,
                              void* d_out, int out_size, void* d_ws, size_t ws_size,
                              hipStream_t stream)
{
    const float* logits = (const float*)d_in[0];
    const int*   labels = (const int*)d_in[1];
    const int    N      = in_sizes[1];          // labels: one per row
    float*       out    = (float*)d_out;
    float*       ws     = (float*)d_ws;

    // Each block writes 32 floats (30 used) of partials into d_ws.
    int nblocks = 2048;
    const int maxBlocksWs = (int)(ws_size / (32 * sizeof(float)));
    if (nblocks > maxBlocksWs) nblocks = maxBlocksWs;
    if (nblocks < 1) nblocks = 1;

    ece_main<<<nblocks, 256, 0, stream>>>(logits, labels, ws, N);
    ece_finalize<<<1, 32, 0, stream>>>(ws, nblocks, 1.0f / (float)N, out);
}
